// GS3DRenderer_49512382988492
// MI455X (gfx1250) — matrix operand order repaired and checked by reading, not compiled, not run on hardware
//
#include <hip/hip_runtime.h>
#include <hip/hip_bf16.h>
#include <math.h>

typedef __attribute__((ext_vector_type(16))) _Float16 v16h;
typedef __attribute__((ext_vector_type(8)))  float    v8f;
typedef __attribute__((ext_vector_type(4)))  unsigned int u32x4;
typedef __attribute__((ext_vector_type(8)))  int          i32x8;
typedef __attribute__((ext_vector_type(4)))  int          i32x4;

#define N_GAUSS 2048
#define FEAT    128
#define IMG_W   128
#define IMG_H   128
#define N_PIX   (IMG_W * IMG_H)
#define TBL_ELEMS (9 * N_GAUSS)              // 9 sorted SoA fields
#define TBL_BYTES (TBL_ELEMS * 4)            // 73728 B of LDS per workgroup

__device__ __forceinline__ float sigmoidf_(float x) {
    return 1.0f / (1.0f + __expf(-x));
}

// ---------------------------------------------------------------------------
// Kernel 1: MLP head via WMMA.  heads[g][0..13] = x[g] . W[col]  (no bias).
// One wave per 16 gaussians. K=128 -> 4 x v_wmma_f32_16x16x32_f16 (unrolled).
// N columns: 0-2 shs, 3-5 scale, 6-8 xyz, 9 opac, 10-13 rot, 14-15 zero pad.
// ---------------------------------------------------------------------------
__global__ __launch_bounds__(32) void mlp_wmma_kernel(
    const float* __restrict__ x,
    const float* __restrict__ w_shs, const float* __restrict__ w_scale,
    const float* __restrict__ w_xyz, const float* __restrict__ w_opac,
    const float* __restrict__ w_rot,
    float* __restrict__ heads)
{
    const int lane = threadIdx.x;          // 0..31
    const int g0   = blockIdx.x * 16;
    const int h    = lane >> 4;            // half-wave
    const int col  = lane & 15;            // N column / A row (M)

    // pick weight row for this lane's N column (B matrix column)
    const float* wrow = w_shs;             // safe fallback for padded cols
    if      (col < 3)   wrow = w_shs   + col * FEAT;
    else if (col < 6)   wrow = w_scale + (col - 3) * FEAT;
    else if (col < 9)   wrow = w_xyz   + (col - 6) * FEAT;
    else if (col == 9)  wrow = w_opac;
    else if (col < 14)  wrow = w_rot   + (col - 10) * FEAT;
    const bool valid_col = (col < 14);

    const float* xrow = x + (size_t)(g0 + col) * FEAT;  // A row M = lane&15

    v8f acc = {};
#pragma unroll
    for (int kc = 0; kc < 4; ++kc) {
        const int kb = kc * 32;
        v16h A, B;
#pragma unroll
        for (int j = 0; j < 16; ++j) {
            // 16-bit A 16x32 layout: K = (j/8)*16 + 8h + (j&7)
            const int KA = kb + ((j >> 3) << 4) + (h << 3) + (j & 7);
            A[j] = (_Float16)xrow[KA];
            // 16-bit B 32x16 layout: K = (j/8)*16 + 8h + (j&7)
            const int KB = kb + ((j >> 3) << 4) + (h << 3) + (j & 7);
            const float wv = wrow[KB];
            B[j] = valid_col ? (_Float16)wv : (_Float16)0.0f;
        }
        acc = __builtin_amdgcn_wmma_f32_16x16x32_f16(
            false, A, false, B, (short)0, acc, false, false);
    }
    // D layout: lane holds N=col, rows M = r + 8h
#pragma unroll
    for (int r = 0; r < 8; ++r)
        heads[(size_t)(g0 + r + 8 * h) * 16 + col] = acc[r];
}

// ---------------------------------------------------------------------------
// Kernel 2: per-gaussian geometry -> params SoA P[f*2048 + g]
// f: 0 u, 1 v, 2 cA, 3 cB, 4 cC, 5 opac*vis, 6 r, 7 g, 8 b, 9 tz(sort key)
// ---------------------------------------------------------------------------
__global__ __launch_bounds__(256) void geom_kernel(
    const float* __restrict__ heads, const float* __restrict__ pts,
    const float* __restrict__ vm,
    const float* __restrict__ b_shs, const float* __restrict__ b_scale,
    const float* __restrict__ b_xyz, const float* __restrict__ b_opac,
    const float* __restrict__ b_rot,
    float* __restrict__ P)
{
    const int g = blockIdx.x * 256 + threadIdx.x;
    if (g >= N_GAUSS) return;
    const float* hd = heads + (size_t)g * 16;

    float rgb[3], s2[3], xyz[3];
#pragma unroll
    for (int i = 0; i < 3; ++i) rgb[i] = sigmoidf_(hd[i] + b_shs[i]);
#pragma unroll
    for (int i = 0; i < 3; ++i) {
        float sc = fminf(__expf(hd[3 + i] + b_scale[i]), 0.2f); // clip [0,0.2]
        s2[i] = sc * sc;
    }
#pragma unroll
    for (int i = 0; i < 3; ++i) {
        float off = (sigmoidf_(hd[6 + i] + b_xyz[i]) - 0.5f) * 0.05f;
        xyz[i] = pts[g * 3 + i] + off;
    }
    const float opac = sigmoidf_(hd[9] + b_opac[0]);

    float q[4];
#pragma unroll
    for (int i = 0; i < 4; ++i) q[i] = hd[10 + i] + b_rot[i];
    const float qn = 1.0f / sqrtf(q[0]*q[0] + q[1]*q[1] + q[2]*q[2] + q[3]*q[3]);
    const float qw = q[0]*qn, qx = q[1]*qn, qy = q[2]*qn, qz = q[3]*qn;

    float R[3][3];
    R[0][0] = 1.f - 2.f*(qy*qy + qz*qz); R[0][1] = 2.f*(qx*qy - qw*qz); R[0][2] = 2.f*(qx*qz + qw*qy);
    R[1][0] = 2.f*(qx*qy + qw*qz); R[1][1] = 1.f - 2.f*(qx*qx + qz*qz); R[1][2] = 2.f*(qy*qz - qw*qx);
    R[2][0] = 2.f*(qx*qz - qw*qy); R[2][1] = 2.f*(qy*qz + qw*qx); R[2][2] = 1.f - 2.f*(qx*qx + qy*qy);

    float cov[3][3];
#pragma unroll
    for (int i = 0; i < 3; ++i)
#pragma unroll
        for (int j = 0; j < 3; ++j)
            cov[i][j] = R[i][0]*s2[0]*R[j][0] + R[i][1]*s2[1]*R[j][1] + R[i][2]*s2[2]*R[j][2];

    float Rv[3][3], tv[3];
#pragma unroll
    for (int i = 0; i < 3; ++i) {
#pragma unroll
        for (int j = 0; j < 3; ++j) Rv[i][j] = vm[i * 4 + j];
        tv[i] = vm[i * 4 + 3];
    }
    float p3[3];
#pragma unroll
    for (int i = 0; i < 3; ++i)
        p3[i] = Rv[i][0]*xyz[0] + Rv[i][1]*xyz[1] + Rv[i][2]*xyz[2] + tv[i];

    const float tz   = fmaxf(p3[2], 0.001f);
    const float itz  = 1.0f / tz;
    const float u    = 128.0f * p3[0] * itz + 64.0f;
    const float v    = 128.0f * p3[1] * itz + 64.0f;

    float J[2][3];
    J[0][0] = 128.0f * itz; J[0][1] = 0.0f;          J[0][2] = -128.0f * p3[0] * itz * itz;
    J[1][0] = 0.0f;          J[1][1] = 128.0f * itz; J[1][2] = -128.0f * p3[1] * itz * itz;

    float M[2][3];
#pragma unroll
    for (int i = 0; i < 2; ++i)
#pragma unroll
        for (int j = 0; j < 3; ++j)
            M[i][j] = J[i][0]*Rv[0][j] + J[i][1]*Rv[1][j] + J[i][2]*Rv[2][j];

    float t2[2][3];
#pragma unroll
    for (int i = 0; i < 2; ++i)
#pragma unroll
        for (int a = 0; a < 3; ++a)
            t2[i][a] = M[i][0]*cov[0][a] + M[i][1]*cov[1][a] + M[i][2]*cov[2][a];

    const float C00 = t2[0][0]*M[0][0] + t2[0][1]*M[0][1] + t2[0][2]*M[0][2];
    const float C01 = t2[0][0]*M[1][0] + t2[0][1]*M[1][1] + t2[0][2]*M[1][2];
    const float C11 = t2[1][0]*M[1][0] + t2[1][1]*M[1][1] + t2[1][2]*M[1][2];

    const float a  = C00 + 0.3f, bb = C01, c = C11 + 0.3f;
    const float det = a * c - bb * bb + 1e-12f;
    const float idet = 1.0f / det;
    const float vis = (p3[2] > 0.01f) ? 1.0f : 0.0f;

    P[0 * N_GAUSS + g] = u;
    P[1 * N_GAUSS + g] = v;
    P[2 * N_GAUSS + g] = c * idet;      // cA
    P[3 * N_GAUSS + g] = -bb * idet;    // cB
    P[4 * N_GAUSS + g] = a * idet;      // cC
    P[5 * N_GAUSS + g] = opac * vis;    // vis folded into opacity
    P[6 * N_GAUSS + g] = rgb[0];
    P[7 * N_GAUSS + g] = rgb[1];
    P[8 * N_GAUSS + g] = rgb[2];
    P[9 * N_GAUSS + g] = tz;            // sort key
}

// ---------------------------------------------------------------------------
// Kernel 3: one-workgroup bitonic argsort (2048 keys, ascending tz) in LDS,
// then gather the 9 param fields into sorted SoA S.
// ---------------------------------------------------------------------------
__global__ __launch_bounds__(1024) void sort_gather_kernel(
    const float* __restrict__ P, float* __restrict__ S)
{
    __shared__ float k_s[N_GAUSS];
    __shared__ int   i_s[N_GAUSS];
    const int t = threadIdx.x;

    for (int i = t; i < N_GAUSS; i += 1024) {
        k_s[i] = P[9 * N_GAUSS + i];
        i_s[i] = i;
    }
    __syncthreads();

    for (int kk = 2; kk <= N_GAUSS; kk <<= 1) {
        for (int jj = kk >> 1; jj > 0; jj >>= 1) {
            const int i   = ((t / jj) * (jj << 1)) + (t % jj);
            const int ixj = i + jj;
            const float a = k_s[i], b = k_s[ixj];
            const bool up = ((i & kk) == 0);
            if (up ? (a > b) : (a < b)) {
                k_s[i] = b; k_s[ixj] = a;
                const int ia = i_s[i]; i_s[i] = i_s[ixj]; i_s[ixj] = ia;
            }
            __syncthreads();
        }
    }
    for (int i = t; i < N_GAUSS; i += 1024) {
        const int src = i_s[i];
#pragma unroll
        for (int f = 0; f < 9; ++f)
            S[f * N_GAUSS + i] = P[f * N_GAUSS + src];
    }
}

// ---------------------------------------------------------------------------
// Kernel 4: splat + front-to-back composite. 16 pixels per wave, 32 sorted
// gaussians per chunk. The 72KB sorted table is DMA'd into LDS once per
// workgroup by the Tensor Data Mover (one wave issues the D#, waits on
// TENSORcnt, then the workgroup barrier publishes it); the hot loop is served
// by ds_load broadcasts. Power/alpha in f32 VALU (numerically safe); the
// 3-channel weighted accumulation is one v_wmma_f32_16x16x32_f16 per chunk:
//   D(16px x 16[3 rgb cols]) += A(16px x 32g wgt, f16) x B(32g x 16 rgb, f16)
// Transmittance chain: serial 8-runs per lane + one shfl_xor(16) stitch.
// ---------------------------------------------------------------------------
__global__ __launch_bounds__(256) void render_kernel(
    const float* __restrict__ S, float* __restrict__ out)
{
    extern __shared__ float tbl[];   // 9 * 2048 floats, dynamic LDS

#if __has_builtin(__builtin_amdgcn_tensor_load_to_lds)
    if (threadIdx.x < 32) {
        // ---- Tensor DMA descriptor: 1 row x 18432 f32, global -> LDS ----
        const unsigned long long ga = (unsigned long long)(size_t)S;
        const unsigned int lds_base = (unsigned int)(size_t)tbl; // low 32b = LDS offset

        u32x4 g0;
        g0[0] = 1u;                                   // count=1, user mode
        g0[1] = lds_base;                             // lds_addr [63:32]
        g0[2] = (unsigned int)ga;                     // global_addr lo32
        g0[3] = (unsigned int)((ga >> 32) & 0x01FFFFFFull) | (2u << 30); // type=2

        i32x8 g1;
        g1[0] = (int)(2u << 16);                      // wg_mask=0, data_size=4B
        g1[1] = (int)((unsigned)TBL_ELEMS << 16);     // tensor_dim0 lo16 @ [31:16]
        g1[2] = (int)(1u << 16);                      // tensor_dim0 hi=0; tensor_dim1=1
        g1[3] = (int)((unsigned)TBL_ELEMS << 16);     // tile_dim0 = 18432
        g1[4] = 1;                                    // tile_dim1 = 1
        g1[5] = TBL_ELEMS;                            // tensor_dim0_stride lo32
        g1[6] = 0;
        g1[7] = 0;

        i32x4 gz = {};                                // groups 2/3 unused (<=2D)
#if __clang_major__ >= 23
        i32x8 gz8 = {};
        __builtin_amdgcn_tensor_load_to_lds(g0, g1, gz, gz, gz8, 0);
#else
        __builtin_amdgcn_tensor_load_to_lds(g0, g1, gz, gz, 0);
#endif
        __builtin_amdgcn_s_wait_tensorcnt((short)0);
    }
#else
    for (int i = threadIdx.x; i < TBL_ELEMS; i += blockDim.x) tbl[i] = S[i];
#endif
    __syncthreads();

    const float* uS  = tbl + 0 * N_GAUSS;
    const float* vS  = tbl + 1 * N_GAUSS;
    const float* cAS = tbl + 2 * N_GAUSS;
    const float* cBS = tbl + 3 * N_GAUSS;
    const float* cCS = tbl + 4 * N_GAUSS;
    const float* opS = tbl + 5 * N_GAUSS;

    const int lane   = threadIdx.x & 31;
    const int waveId = blockIdx.x * (blockDim.x >> 5) + (threadIdx.x >> 5);
    const int h      = lane >> 4;      // half-wave
    const int n      = lane & 15;      // A row (pixel) & D column (channel)

    const int   pix = waveId * 16 + n; // this lane evaluates alpha for pixel n
    const float gx  = (float)(pix & (IMG_W - 1)) + 0.5f;
    const float gy  = (float)(pix >> 7) + 0.5f;

    // this lane's D column: channel n (only 0..2 used)
    const float* rgbp = tbl + (6 + (n < 3 ? n : 0)) * N_GAUSS;
    const bool chan_ok = (n < 3);

    float T = 1.0f;
    v8f img = {};

    for (int cBase = 0; cBase < N_GAUSS; cBase += 32) {
        // ---- alphas for this lane's 16 gaussians (A-matrix K mapping) ----
        float al[16];
#pragma unroll
        for (int j = 0; j < 16; ++j) {
            const int k = cBase + ((j >> 3) << 4) + (h << 3) + (j & 7);
            const float dx = gx - uS[k];
            const float dy = gy - vS[k];
            float pw = -0.5f * (cAS[k] * dx * dx + cCS[k] * dy * dy)
                       - cBS[k] * dx * dy;
            pw = fminf(pw, 0.0f);
            float a = fminf(0.99f, opS[k] * __expf(pw));
            al[j] = (a < (1.0f / 255.0f)) ? 0.0f : a;
        }
        // ---- serial prefix inside each 8-run (local, T excluded) ----
        float w[16];
        float sA = 1.0f, sB = 1.0f;
#pragma unroll
        for (int j = 0; j < 8; ++j) { w[j]     = al[j]     * sA; sA *= (1.0f - al[j]);     }
#pragma unroll
        for (int j = 0; j < 8; ++j) { w[8 + j] = al[8 + j] * sB; sB *= (1.0f - al[8 + j]); }

        // ---- stitch global order across the lane pair sharing pixel n ----
        const float sA_o = __shfl_xor(sA, 16, 32);
        const float sB_o = __shfl_xor(sB, 16, 32);
        const float s0 = h ? sA_o : sA;   // run [base+ 0, +8)
        const float s1 = h ? sA   : sA_o; // run [base+ 8, +8)
        const float s2 = h ? sB_o : sB;   // run [base+16, +8)
        const float s3 = h ? sB   : sB_o; // run [base+24, +8)

        const float Pfirst  = h ? (T * s0)           : T;             // my j<8 run
        const float Psecond = h ? (T * s0 * s1 * s2) : (T * s0 * s1); // my j>=8 run
        T *= s0 * s1 * s2 * s3;

        // ---- build A (weights) and B (rgb) in f16, one WMMA accumulate ----
        v16h A, B;
#pragma unroll
        for (int j = 0; j < 8; ++j) {
            A[j]     = (_Float16)(w[j]     * Pfirst);
            A[8 + j] = (_Float16)(w[8 + j] * Psecond);
        }
#pragma unroll
        for (int j = 0; j < 16; ++j) {
            const int k = cBase + ((j >> 3) << 4) + (h << 3) + (j & 7);  // B layout: K = (j/8)*16 + 8h + (j&7)
            const float rv = rgbp[k];
            B[j] = chan_ok ? (_Float16)rv : (_Float16)0.0f;
        }
        img = __builtin_amdgcn_wmma_f32_16x16x32_f16(
            false, A, false, B, (short)0, img, false, false);
    }

    // D layout: lane holds column n (channel), rows M = r + 8h (pixels)
    if (chan_ok) {
#pragma unroll
        for (int r = 0; r < 8; ++r) {
            const int p = waveId * 16 + r + 8 * h;
            out[(size_t)p * 3 + n] = img[r];
        }
    }
}

// ---------------------------------------------------------------------------
extern "C" void kernel_launch(void* const* d_in, const int* in_sizes, int n_in,
                              void* d_out, int out_size, void* d_ws, size_t ws_size,
                              hipStream_t stream) {
    const float* x       = (const float*)d_in[0];
    const float* pts     = (const float*)d_in[1];
    const float* viewmat = (const float*)d_in[2];
    const float* w_shs   = (const float*)d_in[3];
    const float* b_shs   = (const float*)d_in[4];
    const float* w_scale = (const float*)d_in[5];
    const float* b_scale = (const float*)d_in[6];
    const float* w_xyz   = (const float*)d_in[7];
    const float* b_xyz   = (const float*)d_in[8];
    const float* w_opac  = (const float*)d_in[9];
    const float* b_opac  = (const float*)d_in[10];
    const float* w_rot   = (const float*)d_in[11];
    const float* b_rot   = (const float*)d_in[12];
    float* out = (float*)d_out;

    // workspace partition: heads(128KB) | P unsorted(80KB) | S sorted(72KB)
    float* heads = (float*)d_ws;
    float* P     = heads + (size_t)N_GAUSS * 16;
    float* S     = P + (size_t)10 * N_GAUSS;

    mlp_wmma_kernel<<<N_GAUSS / 16, 32, 0, stream>>>(
        x, w_shs, w_scale, w_xyz, w_opac, w_rot, heads);

    geom_kernel<<<N_GAUSS / 256, 256, 0, stream>>>(
        heads, pts, viewmat, b_shs, b_scale, b_xyz, b_opac, b_rot, P);

    sort_gather_kernel<<<1, 1024, 0, stream>>>(P, S);

    // 16384 pixels / 16 per wave = 1024 waves = 128 blocks x 8 waves
    render_kernel<<<N_PIX / (16 * 8), 256, TBL_BYTES, stream>>>(S, out);
}
